// RelationalAttention_13254269076074
// MI455X (gfx1250) — matrix operand order repaired and checked by reading, not compiled, not run on hardware
//
#include <hip/hip_runtime.h>
#include <cstdint>
#include <cstddef>

// ---------------------------------------------------------------------------
// RelationalAttention, refactored for MI455X (gfx1250, wave32):
//   K1: avgout = mean_c x          (HBM-bound: 411 MB streaming read ~ roofline)
//   K2: Zsum[b][36] = sum_p z_p    (z = relu(bn0(grouped_conv(glcm(avgout)))))
//   K3: single-wave WMMA kernel: pooled = s1*(W1 @ Zmean)+t1 ; att = MLP;
//       veff = (att*s1)^T W1 ; ceff = att . t1      <-- v_wmma_f32_16x16x32_f16
//   K4: out = sigmoid(veff . z_p + ceff)            (L2-resident re-read)
// The spatial mean commutes with the linear 36->36 conv + bn1, so the per-pixel
// GEMM collapses to a per-image GEMM (K3) and a 36-long dot per pixel (K4).
// ---------------------------------------------------------------------------

typedef _Float16 half16  __attribute__((ext_vector_type(16)));
typedef float    float8  __attribute__((ext_vector_type(8)));
typedef float    float4v __attribute__((ext_vector_type(4)));

static constexpr int kB  = 32;
static constexpr int kC  = 64;
static constexpr int kH  = 224;
static constexpr int kW  = 224;
static constexpr int kHW = kH * kW;         // 50176
static constexpr int kN  = 36;
static constexpr int kKP = 64;              // K padded for 16x16x32 f16 WMMA
static constexpr int kNP = 48;              // N padded to 3 tiles of 16

// combinations(range(9),2) flattened -> first/second channel of each pair
__device__ __constant__ int d_pa[36] = {0,0,0,0,0,0,0,0, 1,1,1,1,1,1,1, 2,2,2,2,2,2,
                                        3,3,3,3,3, 4,4,4,4, 5,5,5, 6,6, 7};
__device__ __constant__ int d_pb[36] = {1,2,3,4,5,6,7,8, 2,3,4,5,6,7,8, 3,4,5,6,7,8,
                                        4,5,6,7,8, 5,6,7,8, 6,7,8, 7,8, 8};

// 9 GLCM shift channels of avgout at pixel (ph,pw), torch slice+cat semantics
// (edge rows/cols clamped). ch6 == ch3 analytically.
__device__ __forceinline__ void load9(const float* __restrict__ img, int p, float n[9]) {
    const int ph = p / kW, pw = p - ph * kW;
    const int hp = ph < kH - 1 ? ph + 1 : kH - 1;
    const int hm = ph > 0      ? ph - 1 : 0;
    const int wp = pw < kW - 1 ? pw + 1 : kW - 1;
    const int wm = pw > 0      ? pw - 1 : 0;
    const float* rc = img + ph * kW;
    const float* rp = img + hp * kW;
    const float* rm = img + hm * kW;
    n[0] = rc[pw];  n[1] = rc[wp];  n[2] = rp[pw];  n[3] = rp[wp];  n[4] = rc[wm];
    n[5] = rm[pw];  n[6] = rp[wp];  n[7] = rm[wp];  n[8] = rm[wm];
}

// --------------------------- K0: zero accumulators -------------------------
__global__ void k0_zero(float* __restrict__ Zsum) {
    int i = blockIdx.x * 256 + threadIdx.x;
    if (i < kB * kN) Zsum[i] = 0.0f;
}

// --------------------- K1: channel mean (411 MB stream) ---------------------
__global__ __launch_bounds__(256) void k1_mean64(const float* __restrict__ x,
                                                 float* __restrict__ avg) {
    const int  QPI = kHW / 4;                        // float4 quads per image plane
    const int  q   = blockIdx.x * 256 + threadIdx.x; // 0 .. kB*QPI-1
    const int  b   = q / QPI;
    const int  r   = q - b * QPI;
    const float4v* xp = reinterpret_cast<const float4v*>(x) + (size_t)b * kC * QPI + r;
    float4v acc = {0.f, 0.f, 0.f, 0.f};
#pragma unroll 8
    for (int c = 0; c < kC; ++c)
        acc += __builtin_nontemporal_load(xp + (size_t)c * QPI);   // NT: don't thrash L2
    acc *= (1.0f / 64.0f);
    reinterpret_cast<float4v*>(avg)[q] = acc;
}

// ------------- K2: Zsum[b][36] = sum over pixels of z (fused GLCM) ----------
// 14 blocks per image, 256 threads, 14 pixels/thread; register accumulators,
// wave32 butterfly reduce, one global f32 atomic per (block, channel).
__global__ __launch_bounds__(256) void k2_zsum(const float* __restrict__ avg,
                                               const float* __restrict__ gw,
                                               const float* __restrict__ bng_g,
                                               const float* __restrict__ bng_b,
                                               const float* __restrict__ bng_m,
                                               const float* __restrict__ bng_v,
                                               float* __restrict__ Zsum) {
    __shared__ float A0[kN], A1[kN], T0[kN];
    __shared__ float sZw[8][kN];
    const int t = threadIdx.x;
    if (t < kN) {
        float s = bng_g[t] * rsqrtf(bng_v[t] + 1e-5f);
        A0[t] = s * gw[2 * t];
        A1[t] = s * gw[2 * t + 1];
        T0[t] = bng_b[t] - bng_m[t] * s;
    }
    __syncthreads();

    const int BPI   = 14;                       // blocks per image
    const int b     = blockIdx.x / BPI;
    const int chunk = blockIdx.x - b * BPI;     // 3584 pixels per block
    const float* img = avg + (size_t)b * kHW;

    float zacc[kN];
#pragma unroll
    for (int g = 0; g < kN; ++g) zacc[g] = 0.0f;

    for (int i = 0; i < 14; ++i) {
        const int p = chunk * 3584 + i * 256 + t;
        float n[9];
        load9(img, p, n);
#pragma unroll
        for (int g = 0; g < kN; ++g) {
            float z = fmaf(A0[g], n[d_pa[g]], fmaf(A1[g], n[d_pb[g]], T0[g]));
            zacc[g] += fmaxf(z, 0.0f);
        }
    }

    const int lane = t & 31, w = t >> 5;
#pragma unroll
    for (int g = 0; g < kN; ++g) {
        float v = zacc[g];
#pragma unroll
        for (int off = 16; off; off >>= 1) v += __shfl_xor(v, off, 32);
        if (lane == (g & 31)) sZw[w][g] = v;    // one lane per channel holds wave sum
    }
    __syncthreads();
    if (t < kN) {
        float s = 0.0f;
#pragma unroll
        for (int ww = 0; ww < 8; ++ww) s += sZw[ww][t];
        unsafeAtomicAdd(&Zsum[b * kN + t], s);  // global_atomic_add_f32
    }
}

// --------- K3: single-wave WMMA kernel: pooled -> att -> veff/ceff ----------
// Operands staged in zero-padded f16 LDS so fragment builds are unconditional
// contiguous ds_load_b128 runs (no per-element predication).
// A-frag (f16 16x32, ISA layout): row M=lane%16; slots 0..7 -> K=(hi?8:0)+s,
//   slots 8..15 -> K=(hi?24:16)+(s-8)   (+32 per k-step)
// B-frag (f16 32x16): column N=lane%16; same K slots as A-frag (+32 per k-step)
// C/D (f32 16x16): VGPR r -> (M = r + (hi?8:0), N = lane%16)

__device__ __forceinline__ half16 frag_a(const _Float16* __restrict__ row,
                                         bool hi, int ks) {
    half16 a;
    const int k0 = (hi ? 8 : 0) + ks * 32;
    const int k1 = (hi ? 24 : 16) + ks * 32;
#pragma unroll
    for (int s = 0; s < 8; ++s) { a[s] = row[k0 + s]; a[8 + s] = row[k1 + s]; }
    return a;
}
__device__ __forceinline__ half16 frag_b(const _Float16* __restrict__ col,
                                         bool hi, int ks) {
    half16 b;
    const int k0 = (hi ? 8 : 0) + ks * 32;
    const int k1 = (hi ? 24 : 16) + ks * 32;
#pragma unroll
    for (int s = 0; s < 8; ++s) { b[s] = col[k0 + s]; b[8 + s] = col[k1 + s]; }
    return b;
}

__global__ __launch_bounds__(32) void k3_attention(const float* __restrict__ Zsum,
                                                   const float* __restrict__ w1,
                                                   const float* __restrict__ ca_w1,
                                                   const float* __restrict__ ca_w2,
                                                   const float* __restrict__ bn1_g,
                                                   const float* __restrict__ bn1_b,
                                                   const float* __restrict__ bn1_m,
                                                   const float* __restrict__ bn1_v,
                                                   float* __restrict__ veff,
                                                   float* __restrict__ ceff) {
    __shared__ _Float16 sZh[kB][kKP];    // Zmean (f16), zero-padded K
    __shared__ _Float16 sB1[kNP][kKP];   // GEMM1 B: [o][c] = w1[o][c]
    __shared__ _Float16 sB2[kNP][kKP];   // GEMM2 B: [c][o] = w1[o][c] (transposed)
    __shared__ _Float16 sAh[kB][kKP];    // att*s1 (f16), zero-padded K
    __shared__ float    sS1[kNP], sT1[kNP];
    __shared__ float    sPool[kB][kNP];
    const int t  = threadIdx.x;
    const bool hi = t >= 16;
    const int lm = t & 15;

    // zero-fill padded staging arrays (32 lanes, strided)
    for (int i = t; i < kB * kKP; i += 32)  { sZh[0][i] = (_Float16)0.0f; sAh[0][i] = (_Float16)0.0f; }
    for (int i = t; i < kNP * kKP; i += 32) { sB1[0][i] = (_Float16)0.0f; sB2[0][i] = (_Float16)0.0f; }
    for (int i = t; i < kNP; i += 32)       { sS1[i] = 0.0f; sT1[i] = 0.0f; }
    __syncthreads();

    for (int c = 0; c < kN; ++c)
        sZh[t][c] = (_Float16)(Zsum[t * kN + c] * (1.0f / (float)kHW));
    for (int i = t; i < kN * kN; i += 32) {
        const int o = i / kN, c = i - o * kN;
        const _Float16 v = (_Float16)w1[i];
        sB1[o][c] = v;
        sB2[c][o] = v;
    }
    for (int i = t; i < kN; i += 32) {
        float s = bn1_g[i] * rsqrtf(bn1_v[i] + 1e-5f);
        sS1[i] = s;
        sT1[i] = bn1_b[i] - bn1_m[i] * s;
    }
    __syncthreads();

    // GEMM1: pooled[b][o] = s1[o] * sum_c Zmean[b][c]*w1[o][c] + t1[o]
#pragma unroll
    for (int mt = 0; mt < 2; ++mt)
#pragma unroll
        for (int nt = 0; nt < 3; ++nt) {
            float8 acc = {0, 0, 0, 0, 0, 0, 0, 0};
#pragma unroll
            for (int ks = 0; ks < 2; ++ks) {
                half16 a = frag_a(&sZh[mt * 16 + lm][0], hi, ks);
                half16 b = frag_b(&sB1[nt * 16 + lm][0], hi, ks);
                acc = __builtin_amdgcn_wmma_f32_16x16x32_f16(false, a, false, b,
                                                             (short)0, acc, false, false);
            }
#pragma unroll
            for (int r = 0; r < 8; ++r) {
                const int m = mt * 16 + r + (hi ? 8 : 0);
                const int o = nt * 16 + lm;
                sPool[m][o] = sS1[o] * acc[r] + sT1[o];     // padded: unconditional
            }
        }
    __syncthreads();

    // channel-attention MLP, one batch per lane (B == 32 == wave32)
    {
        float hd[18];
#pragma unroll
        for (int h = 0; h < 18; ++h) {
            float s = 0.0f;
            for (int o = 0; o < kN; ++o) s += sPool[t][o] * ca_w1[h * kN + o];
            hd[h] = fmaxf(s, 0.0f);
        }
        float ce = 0.0f;
        for (int o = 0; o < kN; ++o) {
            float s = 0.0f;
#pragma unroll
            for (int h = 0; h < 18; ++h) s += hd[h] * ca_w2[o * 18 + h];
            const float att = 1.0f / (1.0f + __expf(-s));
            sAh[t][o] = (_Float16)(att * sS1[o]);
            ce += att * sT1[o];
        }
        ceff[t] = ce;
    }
    __syncthreads();

    // GEMM2: veff[b][c] = sum_o (att*s1)[b][o] * w1[o][c]
#pragma unroll
    for (int mt = 0; mt < 2; ++mt)
#pragma unroll
        for (int nt = 0; nt < 3; ++nt) {
            float8 acc = {0, 0, 0, 0, 0, 0, 0, 0};
#pragma unroll
            for (int ks = 0; ks < 2; ++ks) {
                half16 a = frag_a(&sAh[mt * 16 + lm][0], hi, ks);
                half16 b = frag_b(&sB2[nt * 16 + lm][0], hi, ks);
                acc = __builtin_amdgcn_wmma_f32_16x16x32_f16(false, a, false, b,
                                                             (short)0, acc, false, false);
            }
#pragma unroll
            for (int r = 0; r < 8; ++r) {
                const int m = mt * 16 + r + (hi ? 8 : 0);
                const int c = nt * 16 + lm;
                if (c < kN) veff[m * kN + c] = acc[r];      // exec-masked store only
            }
        }
}

// ---------------- K4: out = sigmoid(veff . z + ceff), L2-resident ----------
__global__ __launch_bounds__(256) void k4_out(const float* __restrict__ avg,
                                              const float* __restrict__ gw,
                                              const float* __restrict__ bng_g,
                                              const float* __restrict__ bng_b,
                                              const float* __restrict__ bng_m,
                                              const float* __restrict__ bng_v,
                                              const float* __restrict__ veff,
                                              const float* __restrict__ ceff,
                                              float* __restrict__ out) {
    __shared__ float A0[kN], A1[kN], T0[kN], V[kN];
    __shared__ float CE;
    const int t   = threadIdx.x;
    const int BPI = kHW / 256;                 // 196 blocks per image
    const int b   = blockIdx.x / BPI;
    if (t < kN) {
        float s = bng_g[t] * rsqrtf(bng_v[t] + 1e-5f);
        A0[t] = s * gw[2 * t];
        A1[t] = s * gw[2 * t + 1];
        T0[t] = bng_b[t] - bng_m[t] * s;
        V[t]  = veff[b * kN + t];
    }
    if (t == 0) CE = ceff[b];
    __syncthreads();

    const int p = (blockIdx.x - b * BPI) * 256 + t;
    const float* img = avg + (size_t)b * kHW;
    float n[9];
    load9(img, p, n);
    float acc = CE;
#pragma unroll
    for (int g = 0; g < kN; ++g) {
        float z = fmaf(A0[g], n[d_pa[g]], fmaf(A1[g], n[d_pb[g]], T0[g]));
        acc = fmaf(V[g], fmaxf(z, 0.0f), acc);
    }
    out[(size_t)b * kHW + p] = 1.0f / (1.0f + __expf(-acc));
}

// ---------------------------------------------------------------------------
extern "C" void kernel_launch(void* const* d_in, const int* in_sizes, int n_in,
                              void* d_out, int out_size, void* d_ws, size_t ws_size,
                              hipStream_t stream) {
    const float* x     = (const float*)d_in[0];
    const float* gw    = (const float*)d_in[1];
    const float* w1    = (const float*)d_in[2];
    const float* ca_w1 = (const float*)d_in[3];
    const float* ca_w2 = (const float*)d_in[4];
    const float* bng_g = (const float*)d_in[5];
    const float* bng_b = (const float*)d_in[6];
    const float* bng_m = (const float*)d_in[7];
    const float* bng_v = (const float*)d_in[8];
    const float* bn1_g = (const float*)d_in[9];
    const float* bn1_b = (const float*)d_in[10];
    const float* bn1_m = (const float*)d_in[11];
    const float* bn1_v = (const float*)d_in[12];
    float* out = (float*)d_out;

    char*  ws   = (char*)d_ws;
    float* avg  = (float*)ws;                                  // kB*kHW floats (6.4 MB)
    float* Zsum = (float*)(ws + (size_t)kB * kHW * sizeof(float));
    float* veff = Zsum + kB * kN;
    float* ceff = veff + kB * kN;

    k0_zero<<<(kB * kN + 255) / 256, 256, 0, stream>>>(Zsum);
    k1_mean64<<<kB * (kHW / 4) / 256, 256, 0, stream>>>(x, avg);
    k2_zsum<<<kB * 14, 256, 0, stream>>>(avg, gw, bng_g, bng_b, bng_m, bng_v, Zsum);
    k3_attention<<<1, 32, 0, stream>>>(Zsum, w1, ca_w1, ca_w2,
                                       bn1_g, bn1_b, bn1_m, bn1_v, veff, ceff);
    k4_out<<<kB * (kHW / 256), 256, 0, stream>>>(avg, gw, bng_g, bng_b, bng_m, bng_v,
                                                 veff, ceff, out);
}
